// WorldModel_4647154614939
// MI455X (gfx1250) — matrix operand order repaired and checked by reading, not compiled, not run on hardware
//
#include <hip/hip_runtime.h>
#include <hip/hip_bf16.h>

// ---------------- problem constants ----------------
#define BB   64
#define TT   128
#define OBS  512
#define HH   1024
#define LL   32
#define CC   32
#define AA   18
#define LC   1024
#define TM1  127
#define ROWS (BB * TM1)          // 8128

// d_out layout (floats): recon | reward | kl | hs | samples
#define RECON_OFF 0
#define REW_OFF   (ROWS * OBS)                 // 4161536
#define KL_OFF    (REW_OFF + ROWS)             // 4169664
#define HS_OFF    (KL_OFF + 1)                 // 4169665
#define SAMP_OFF  (HS_OFF + ROWS * HH)         // 12492737

// ---------------- types / helpers ----------------
typedef __attribute__((ext_vector_type(16))) __bf16 bfrag;
typedef __attribute__((ext_vector_type(8)))  float  f32x8;

union FragU { bfrag v; uint4 q[2]; };

__device__ inline unsigned short f2bf(float f) {
    unsigned int u = __float_as_uint(f);
    unsigned int r = (u + 0x7fffu + ((u >> 16) & 1u)) >> 16;
    return (unsigned short)r;
}

// A-fragment (16x32 bf16, row-major source, lane = row (lane&15)):
//   v0..v3 = K[kb..kb+8), v4..v7 = K[kb+16..kb+24), kb = (lane>>4)*8
__device__ inline bfrag load_a(const unsigned short* p /* row base + k0 */, int lh) {
    FragU f;
    f.q[0] = *(const uint4*)(p + lh * 8);
    f.q[1] = *(const uint4*)(p + lh * 8 + 16);
    return f.v;
}

// B-fragment (32x16 bf16, N-major source, lane = col (lane&15)):
//   v0..v3 = K[kb..kb+8), v4..v7 = K[kb+16..kb+24), kb = (lane>>4)*8
__device__ inline bfrag load_b(const unsigned short* p /* col base + k0 */, int lh) {
    FragU f;
    f.q[0] = *(const uint4*)(p + lh * 8);
    f.q[1] = *(const uint4*)(p + lh * 8 + 16);
    return f.v;
}

__device__ inline f32x8 wmma_bf(bfrag a, bfrag b, f32x8 c) {
    return __builtin_amdgcn_wmma_f32_16x16x32_bf16(false, a, false, b, (short)0, c, false, false);
}

__device__ inline float sigm(float x) { return 1.0f / (1.0f + __expf(-x)); }

// ---------------- conversion kernels ----------------
__global__ void k_cvt(const float* __restrict__ src, unsigned short* __restrict__ dst, int n) {
    int i = blockIdx.x * blockDim.x + threadIdx.x;
    if (i < n) dst[i] = f2bf(src[i]);
}

// copy first dst_cols of each row of a (rows x src_stride) matrix
__global__ void k_cvt_rows(const float* __restrict__ src, unsigned short* __restrict__ dst,
                           int src_stride, int dst_cols, int n) {
    int i = blockIdx.x * blockDim.x + threadIdx.x;
    if (i >= n) return;
    int r = i / dst_cols, c = i - r * dst_cols;
    dst[i] = f2bf(src[(long)r * src_stride + c]);
}

// dst[c*R + r] = src[r*C + c]   (src R x C  ->  dst C x R, bf16)
__global__ void k_transpose(const float* __restrict__ src, unsigned short* __restrict__ dst,
                            int R, int Ccols) {
    long i = (long)blockIdx.x * blockDim.x + threadIdx.x;
    long n = (long)R * Ccols;
    if (i >= n) return;
    long c = i / R, r = i - c * R;
    dst[c * R + r] = f2bf(src[r * Ccols + c]);
}

__global__ void k_zero1(float* p) { *p = 0.0f; }

// ---------------- encoder: e = relu(relu(obs@W1+b1)@W2+b2), bf16 out ------------
__global__ void __launch_bounds__(512)
k_encoder(const unsigned short* __restrict__ obs_bf, const float* __restrict__ b1,
          const float* __restrict__ b2, const unsigned short* __restrict__ w1T,
          const unsigned short* __restrict__ w2T, unsigned short* __restrict__ e_bf) {
    extern __shared__ unsigned char dynsm[];
    unsigned short* h1 = (unsigned short*)dynsm;  // 16 x 1024 bf16
    const int tid = threadIdx.x, lane = tid & 31, wv = tid >> 5;
    const int lh = lane >> 4, ln = lane & 15;
    const long row0 = (long)blockIdx.x * 16;

    for (int i = 0; i < 4; ++i) {
        int c0 = (wv * 4 + i) * 16;
        f32x8 acc = {};
        const unsigned short* ar = obs_bf + (row0 + ln) * OBS;
        const unsigned short* bc = w1T + (long)(c0 + ln) * OBS;
        for (int k0 = 0; k0 < OBS; k0 += 32)
            acc = wmma_bf(load_a(ar + k0, lh), load_b(bc + k0, lh), acc);
        int col = c0 + ln;
        float bias = b1[col];
        for (int ei = 0; ei < 8; ++ei) {
            int m = ei + 8 * lh;
            float v = acc[ei] + bias;
            h1[m * HH + col] = f2bf(v > 0.f ? v : 0.f);
        }
    }
    __syncthreads();
    for (int i = 0; i < 4; ++i) {
        int c0 = (wv * 4 + i) * 16;
        f32x8 acc = {};
        const unsigned short* ar = h1 + ln * HH;
        const unsigned short* bc = w2T + (long)(c0 + ln) * HH;
        for (int k0 = 0; k0 < HH; k0 += 32)
            acc = wmma_bf(load_a(ar + k0, lh), load_b(bc + k0, lh), acc);
        int col = c0 + ln;
        float bias = b2[col];
        for (int ei = 0; ei < 8; ++ei) {
            int m = ei + 8 * lh;
            float v = acc[ei] + bias;
            e_bf[(row0 + m) * HH + col] = f2bf(v > 0.f ? v : 0.f);
        }
    }
}

// ---------------- GRU scan (persistent, 1 workgroup per 16-row batch tile) -------
__global__ void __launch_bounds__(512)
k_gru(const unsigned short* __restrict__ e_bf, const unsigned short* __restrict__ wih,
      const unsigned short* __restrict__ whh, const float* __restrict__ wih_f32,
      const float* __restrict__ bih, const float* __restrict__ bhh,
      const unsigned short* __restrict__ postT, const float* __restrict__ post_b,
      const float* __restrict__ gumbel, const int* __restrict__ act_seq,
      float* __restrict__ out_hs, float* __restrict__ out_samples,
      unsigned short* __restrict__ hs_bf, unsigned short* __restrict__ samples_bf,
      float* __restrict__ post_logits) {
    extern __shared__ unsigned char dynsm[];
    float*          h_f   = (float*)dynsm;                                      // [2][16][HH] f32
    unsigned short* h_bf  = (unsigned short*)(dynsm + 2 * 16 * HH * 4);         // [2][16][HH] bf16
    unsigned short* samp  = (unsigned short*)(dynsm + 2 * 16 * HH * 4 + 2 * 16 * HH * 2); // [16][HH]
    float*          logit = (float*)(dynsm + 2 * 16 * HH * 4 + 3 * 16 * HH * 2);          // [16][HH]

    const int tid = threadIdx.x, lane = tid & 31, wv = tid >> 5;
    const int lh = lane >> 4, ln = lane & 15;
    const int b0 = blockIdx.x * 16;

    for (int i = tid; i < 2 * 16 * HH; i += 512) { h_f[i] = 0.f; h_bf[i] = 0; }
    __syncthreads();

    // ----- initial posterior: logits0 = [0, e[:,0]] @ postT + b -----
    for (int it = 0; it < 4; ++it) {
        int c0 = (wv + it * 16) * 16;
        f32x8 acc = {};
        const unsigned short* ah = h_bf + ln * HH;  // zeros
        const unsigned short* bc = postT + (long)(c0 + ln) * (2 * HH);
        for (int k0 = 0; k0 < HH; k0 += 32)
            acc = wmma_bf(load_a(ah + k0, lh), load_b(bc + k0, lh), acc);
        const unsigned short* ae = e_bf + ((long)(b0 + ln) * TT) * HH;
        for (int k0 = 0; k0 < HH; k0 += 32)
            acc = wmma_bf(load_a(ae + k0, lh), load_b(bc + HH + k0, lh), acc);
        int col = c0 + ln;
        float pb = post_b[col];
        for (int ei = 0; ei < 8; ++ei) logit[(ei + 8 * lh) * HH + col] = acc[ei] + pb;
    }
    __syncthreads();
    {   // samp0 = one_hot(argmax(logits0 + gumbel[0])); samples row 0
        int m = tid >> 5, l = tid & 31;
        const float* lg = logit + m * HH + l * CC;
        const float* gg = gumbel + ((long)(b0 + m)) * LC + l * CC;
        int amax = 0; float best = -3.4e38f;
        for (int c = 0; c < CC; ++c) { float v = lg[c] + gg[c]; if (v > best) { best = v; amax = c; } }
        long orow = (long)(b0 + m) * TM1;
        for (int c = 0; c < CC; ++c) {
            float v = (c == amax) ? 1.f : 0.f;
            samp[m * HH + l * CC + c] = f2bf(v);
            out_samples[orow * LC + l * CC + c] = v;
            samples_bf[orow * LC + l * CC + c] = f2bf(v);
        }
    }
    __syncthreads();

    int cur = 0;
    for (int t = 0; t < TM1; ++t) {
        int nxt = cur ^ 1;
        // ----- GRU cell: column tiles of H -----
        for (int it = 0; it < 4; ++it) {
            int j0 = (wv + it * 16) * 16;
            f32x8 air = {}, aiz = {}, ain = {}, ahr = {}, ahz = {}, ahn = {};
            const unsigned short* as = samp + ln * HH;
            const unsigned short* ah = h_bf + cur * 16 * HH + ln * HH;
            const unsigned short* bir = wih + (long)(0 * HH + j0 + ln) * HH;
            const unsigned short* biz = wih + (long)(1 * HH + j0 + ln) * HH;
            const unsigned short* bin = wih + (long)(2 * HH + j0 + ln) * HH;
            const unsigned short* bhr = whh + (long)(0 * HH + j0 + ln) * HH;
            const unsigned short* bhz = whh + (long)(1 * HH + j0 + ln) * HH;
            const unsigned short* bhn = whh + (long)(2 * HH + j0 + ln) * HH;
            for (int k0 = 0; k0 < HH; k0 += 32) {
                bfrag fa = load_a(as + k0, lh);
                bfrag fh = load_a(ah + k0, lh);
                air = wmma_bf(fa, load_b(bir + k0, lh), air);
                aiz = wmma_bf(fa, load_b(biz + k0, lh), aiz);
                ain = wmma_bf(fa, load_b(bin + k0, lh), ain);
                ahr = wmma_bf(fh, load_b(bhr + k0, lh), ahr);
                ahz = wmma_bf(fh, load_b(bhz + k0, lh), ahz);
                ahn = wmma_bf(fh, load_b(bhn + k0, lh), ahn);
            }
            int j = j0 + ln;
            float b_ir = bih[j], b_iz = bih[HH + j], b_in = bih[2 * HH + j];
            float c_hr = bhh[j], c_hz = bhh[HH + j], c_hn = bhh[2 * HH + j];
            for (int ei = 0; ei < 8; ++ei) {
                int m = ei + 8 * lh;
                int a = act_seq[(b0 + m) * TT + t];               // one-hot column gather
                float ir = air[ei] + b_ir + wih_f32[(long)(0 * HH + j) * (LC + AA) + LC + a];
                float iz = aiz[ei] + b_iz + wih_f32[(long)(1 * HH + j) * (LC + AA) + LC + a];
                float in_ = ain[ei] + b_in + wih_f32[(long)(2 * HH + j) * (LC + AA) + LC + a];
                float r = sigm(ir + ahr[ei] + c_hr);
                float z = sigm(iz + ahz[ei] + c_hz);
                float n = tanhf(in_ + r * (ahn[ei] + c_hn));
                float ho = h_f[cur * 16 * HH + m * HH + j];
                float hv = (1.f - z) * n + z * ho;
                h_f[nxt * 16 * HH + m * HH + j] = hv;
                h_bf[nxt * 16 * HH + m * HH + j] = f2bf(hv);
                long orow = (long)(b0 + m) * TM1 + t;
                out_hs[orow * HH + j] = hv;
                hs_bf[orow * HH + j] = f2bf(hv);
            }
        }
        __syncthreads();
        // ----- posterior: logits = [h_new, e[:,t+1]] @ postT + b (== post_logits[t]) -----
        for (int it = 0; it < 4; ++it) {
            int c0 = (wv + it * 16) * 16;
            f32x8 acc = {};
            const unsigned short* ah = h_bf + nxt * 16 * HH + ln * HH;
            const unsigned short* bc = postT + (long)(c0 + ln) * (2 * HH);
            for (int k0 = 0; k0 < HH; k0 += 32)
                acc = wmma_bf(load_a(ah + k0, lh), load_b(bc + k0, lh), acc);
            const unsigned short* ae = e_bf + ((long)(b0 + ln) * TT + (t + 1)) * HH;
            for (int k0 = 0; k0 < HH; k0 += 32)
                acc = wmma_bf(load_a(ae + k0, lh), load_b(bc + HH + k0, lh), acc);
            int col = c0 + ln;
            float pb = post_b[col];
            for (int ei = 0; ei < 8; ++ei) {
                int m = ei + 8 * lh;
                float v = acc[ei] + pb;
                logit[m * HH + col] = v;
                post_logits[((long)(b0 + m) * TM1 + t) * LC + col] = v;
            }
        }
        __syncthreads();
        // ----- sample samp_{t+1} (needed only while t < T-2) -----
        if (t < TM1 - 1) {
            int m = tid >> 5, l = tid & 31;
            const float* lg = logit + m * HH + l * CC;
            const float* gg = gumbel + ((long)(t + 1) * BB + (b0 + m)) * LC + l * CC;
            int amax = 0; float best = -3.4e38f;
            for (int c = 0; c < CC; ++c) { float v = lg[c] + gg[c]; if (v > best) { best = v; amax = c; } }
            long orow = (long)(b0 + m) * TM1 + (t + 1);
            for (int c = 0; c < CC; ++c) {
                float v = (c == amax) ? 1.f : 0.f;
                samp[m * HH + l * CC + c] = f2bf(v);
                out_samples[orow * LC + l * CC + c] = v;
                samples_bf[orow * LC + l * CC + c] = f2bf(v);
            }
            __syncthreads();
        }
        cur = nxt;
    }
}

// ---------------- prior logits + KL ----------------
__global__ void __launch_bounds__(512)
k_prior_kl(const unsigned short* __restrict__ hs_bf, const unsigned short* __restrict__ priorT,
           const float* __restrict__ prior_b, const float* __restrict__ post_logits,
           float* __restrict__ out_kl) {
    extern __shared__ unsigned char dynsm[];
    float* pl  = (float*)dynsm;                    // 16 x 1024
    float* red = (float*)(dynsm + 16 * HH * 4);    // 512
    const int tid = threadIdx.x, lane = tid & 31, wv = tid >> 5;
    const int lh = lane >> 4, ln = lane & 15;
    const long row0 = (long)blockIdx.x * 16;

    for (int i = 0; i < 4; ++i) {
        int c0 = (wv * 4 + i) * 16;
        f32x8 acc = {};
        const unsigned short* ar = hs_bf + (row0 + ln) * HH;
        const unsigned short* bc = priorT + (long)(c0 + ln) * HH;
        for (int k0 = 0; k0 < HH; k0 += 32)
            acc = wmma_bf(load_a(ar + k0, lh), load_b(bc + k0, lh), acc);
        int col = c0 + ln;
        float pb = prior_b[col];
        for (int ei = 0; ei < 8; ++ei) pl[(ei + 8 * lh) * HH + col] = acc[ei] + pb;
    }
    __syncthreads();
    {
        int m = tid >> 5, l = tid & 31;
        const float* p = pl + m * HH + l * CC;
        const float* q = post_logits + (row0 + m) * LC + l * CC;
        float pmax = -3.4e38f, qmax = -3.4e38f;
        for (int c = 0; c < CC; ++c) { pmax = fmaxf(pmax, p[c]); qmax = fmaxf(qmax, q[c]); }
        float ps = 0.f, qs = 0.f;
        for (int c = 0; c < CC; ++c) { ps += __expf(p[c] - pmax); qs += __expf(q[c] - qmax); }
        float lps = pmax + __logf(ps), lqs = qmax + __logf(qs);
        float klf = 0.f, klr = 0.f;
        for (int c = 0; c < CC; ++c) {
            float lp = p[c] - lps, lq = q[c] - lqs;
            klf += __expf(lq) * (lq - lp);
            klr += __expf(lp) * (lp - lq);
        }
        red[tid] = 0.8f * klf + 0.2f * klr;
    }
    __syncthreads();
    if (tid < 16) {
        float s = 0.f;
        for (int l = 0; l < CC; ++l) s += red[tid * 32 + l];
        s = fmaxf(s, 1.0f);
        atomicAdd(out_kl, s * (1.0f / (float)ROWS));
    }
}

// ---------------- decoder: recon = relu([hs,samp]@W1+b1)@W2+b2 -------------------
__global__ void __launch_bounds__(512)
k_decoder(const unsigned short* __restrict__ hs_bf, const unsigned short* __restrict__ samples_bf,
          const unsigned short* __restrict__ dec1T, const float* __restrict__ dec_b1,
          const unsigned short* __restrict__ dec2T, const float* __restrict__ dec_b2,
          float* __restrict__ recon) {
    extern __shared__ unsigned char dynsm[];
    unsigned short* h1 = (unsigned short*)dynsm;  // 16 x 1024 bf16
    const int tid = threadIdx.x, lane = tid & 31, wv = tid >> 5;
    const int lh = lane >> 4, ln = lane & 15;
    const long row0 = (long)blockIdx.x * 16;

    for (int i = 0; i < 4; ++i) {
        int c0 = (wv * 4 + i) * 16;
        f32x8 acc = {};
        const unsigned short* ah = hs_bf + (row0 + ln) * HH;
        const unsigned short* as = samples_bf + (row0 + ln) * LC;
        const unsigned short* bc = dec1T + (long)(c0 + ln) * (2 * HH);
        for (int k0 = 0; k0 < HH; k0 += 32)
            acc = wmma_bf(load_a(ah + k0, lh), load_b(bc + k0, lh), acc);
        for (int k0 = 0; k0 < HH; k0 += 32)
            acc = wmma_bf(load_a(as + k0, lh), load_b(bc + HH + k0, lh), acc);
        int col = c0 + ln;
        float b = dec_b1[col];
        for (int ei = 0; ei < 8; ++ei) {
            float v = acc[ei] + b;
            h1[(ei + 8 * lh) * HH + col] = f2bf(v > 0.f ? v : 0.f);
        }
    }
    __syncthreads();
    for (int i = 0; i < 2; ++i) {
        int c0 = (wv * 2 + i) * 16;
        f32x8 acc = {};
        const unsigned short* ar = h1 + ln * HH;
        const unsigned short* bc = dec2T + (long)(c0 + ln) * HH;
        for (int k0 = 0; k0 < HH; k0 += 32)
            acc = wmma_bf(load_a(ar + k0, lh), load_b(bc + k0, lh), acc);
        int col = c0 + ln;
        float b = dec_b2[col];
        for (int ei = 0; ei < 8; ++ei)
            recon[(row0 + ei + 8 * lh) * OBS + col] = acc[ei] + b;
    }
}

// ---------------- reward head (tiny dot per row) ----------------
__global__ void __launch_bounds__(256)
k_reward(const float* __restrict__ hs, const float* __restrict__ samples,
         const float* __restrict__ rw, const float* __restrict__ rb,
         float* __restrict__ out_r) {
    int wv = threadIdx.x >> 5, lane = threadIdx.x & 31;
    long row = (long)blockIdx.x * 8 + wv;
    if (row >= ROWS) return;
    const float* h = hs + row * HH;
    const float* s = samples + row * LC;
    float acc = 0.f;
    for (int k = lane; k < HH; k += 32) acc += h[k] * rw[k] + s[k] * rw[HH + k];
    for (int off = 16; off > 0; off >>= 1) acc += __shfl_down(acc, off, 32);
    if (lane == 0) out_r[row] = acc + rb[0];
}

// ---------------- launcher ----------------
extern "C" void kernel_launch(void* const* d_in, const int* in_sizes, int n_in,
                              void* d_out, int out_size, void* d_ws, size_t ws_size,
                              hipStream_t stream) {
    const float* obs     = (const float*)d_in[0];
    const float* gumbel  = (const float*)d_in[1];
    const float* enc_w1  = (const float*)d_in[2];
    const float* enc_b1  = (const float*)d_in[3];
    const float* enc_w2  = (const float*)d_in[4];
    const float* enc_b2  = (const float*)d_in[5];
    const float* gru_wih = (const float*)d_in[6];
    const float* gru_whh = (const float*)d_in[7];
    const float* gru_bih = (const float*)d_in[8];
    const float* gru_bhh = (const float*)d_in[9];
    const float* prior_w = (const float*)d_in[10];
    const float* prior_b = (const float*)d_in[11];
    const float* post_w  = (const float*)d_in[12];
    const float* post_b  = (const float*)d_in[13];
    const float* dec_w1  = (const float*)d_in[14];
    const float* dec_b1  = (const float*)d_in[15];
    const float* dec_w2  = (const float*)d_in[16];
    const float* dec_b2  = (const float*)d_in[17];
    const float* rew_w   = (const float*)d_in[18];
    const float* rew_b   = (const float*)d_in[19];
    const int*   act     = (const int*)d_in[20];
    float* out = (float*)d_out;

    // workspace carve (256B aligned)
    unsigned char* wsb = (unsigned char*)d_ws;
    size_t off = 0;
    auto take = [&](size_t bytes) -> void* {
        void* p = wsb + off;
        off += (bytes + 255) & ~(size_t)255;
        return p;
    };
    unsigned short* obs_bf     = (unsigned short*)take((size_t)BB * TT * OBS * 2);
    unsigned short* e_bf       = (unsigned short*)take((size_t)BB * TT * HH * 2);
    unsigned short* hs_bf      = (unsigned short*)take((size_t)ROWS * HH * 2);
    unsigned short* samples_bf = (unsigned short*)take((size_t)ROWS * LC * 2);
    float*          plogits    = (float*)take((size_t)ROWS * LC * 4);
    unsigned short* w_enc1T    = (unsigned short*)take((size_t)HH * OBS * 2);
    unsigned short* w_enc2T    = (unsigned short*)take((size_t)HH * HH * 2);
    unsigned short* w_wih      = (unsigned short*)take((size_t)3 * HH * HH * 2);
    unsigned short* w_whh      = (unsigned short*)take((size_t)3 * HH * HH * 2);
    unsigned short* w_priorT   = (unsigned short*)take((size_t)LC * HH * 2);
    unsigned short* w_postT    = (unsigned short*)take((size_t)LC * 2 * HH * 2);
    unsigned short* w_dec1T    = (unsigned short*)take((size_t)HH * 2 * HH * 2);
    unsigned short* w_dec2T    = (unsigned short*)take((size_t)OBS * HH * 2);

    int n;
    n = BB * TT * OBS;
    k_cvt<<<(n + 255) / 256, 256, 0, stream>>>(obs, obs_bf, n);
    n = OBS * HH;
    k_transpose<<<(n + 255) / 256, 256, 0, stream>>>(enc_w1, w_enc1T, OBS, HH);
    n = HH * HH;
    k_transpose<<<(n + 255) / 256, 256, 0, stream>>>(enc_w2, w_enc2T, HH, HH);
    n = 3 * HH * HH;
    k_cvt_rows<<<(n + 255) / 256, 256, 0, stream>>>(gru_wih, w_wih, LC + AA, HH, n);
    k_cvt<<<(n + 255) / 256, 256, 0, stream>>>(gru_whh, w_whh, n);
    n = HH * LC;
    k_transpose<<<(n + 255) / 256, 256, 0, stream>>>(prior_w, w_priorT, HH, LC);
    n = 2 * HH * LC;
    k_transpose<<<(n + 255) / 256, 256, 0, stream>>>(post_w, w_postT, 2 * HH, LC);
    n = 2 * HH * HH;
    k_transpose<<<(n + 255) / 256, 256, 0, stream>>>(dec_w1, w_dec1T, 2 * HH, HH);
    n = HH * OBS;
    k_transpose<<<(n + 255) / 256, 256, 0, stream>>>(dec_w2, w_dec2T, HH, OBS);

    // encoder: 8192 rows / 16 = 512 tiles
    k_encoder<<<512, 512, 16 * HH * 2, stream>>>(obs_bf, enc_b1, enc_b2, w_enc1T, w_enc2T, e_bf);

    k_zero1<<<1, 1, 0, stream>>>(out + KL_OFF);

    // GRU scan: 4 persistent workgroups, 288 KB LDS each
    size_t gru_lds = (size_t)2 * 16 * HH * 4 + (size_t)3 * 16 * HH * 2 + (size_t)16 * HH * 4;
    k_gru<<<4, 512, gru_lds, stream>>>(e_bf, w_wih, w_whh, gru_wih, gru_bih, gru_bhh,
                                       w_postT, post_b, gumbel, act,
                                       out + HS_OFF, out + SAMP_OFF, hs_bf, samples_bf, plogits);

    // prior + KL: 8128 rows / 16 = 508 tiles
    k_prior_kl<<<508, 512, 16 * HH * 4 + 512 * 4, stream>>>(hs_bf, w_priorT, prior_b, plogits,
                                                            out + KL_OFF);
    // decoder
    k_decoder<<<508, 512, 16 * HH * 2, stream>>>(hs_bf, samples_bf, w_dec1T, dec_b1, w_dec2T,
                                                 dec_b2, out + RECON_OFF);
    // reward
    k_reward<<<(ROWS + 7) / 8, 256, 0, stream>>>(out + HS_OFF, out + SAMP_OFF, rew_w, rew_b,
                                                 out + REW_OFF);
}
